// Net_28363964023543
// MI455X (gfx1250) — matrix operand order repaired and checked by reading, not compiled, not run on hardware
//
#include <hip/hip_runtime.h>

typedef __attribute__((ext_vector_type(16))) _Float16 v16h;
typedef __attribute__((ext_vector_type(8)))  _Float16 v8h;
typedef __attribute__((ext_vector_type(8)))  float    v8f;

#define NOUT 192
#define OT_TOTAL 12   // 192 / 16 output-channel tiles

__device__ __forceinline__ float fast_tanh(float x) {
#if defined(__gfx1250__) && __has_builtin(__builtin_amdgcn_tanhf)
    return __builtin_amdgcn_tanhf(x);      // single v_tanh_f32 TRANS op
#else
    // tanh(x) = sign(x) * (1 - 2 e^{-2|x|} / (1 + e^{-2|x|}))
    float a = __builtin_fabsf(x);
    float e = __expf(-2.0f * a);
    float r = 1.0f - 2.0f * e * __builtin_amdgcn_rcpf(1.0f + e);
    return __builtin_copysignf(r, x);
#endif
}

// ---------------------------------------------------------------------------
// Pack f32 weights (O, C, 25) -> f16 WMMA B-fragments, K padded 25->32.
// Layout: dst[((ot*C + c)*32 + lane)*16 + h], value = W[o = ot*16 + (lane&15)]
// at tap t = (lane>>4)*16 + h (contiguous k order, same as A), zero for t >= 25.
// ---------------------------------------------------------------------------
__global__ void pack_wmma_b(const float* __restrict__ src,
                            _Float16* __restrict__ dst, int C) {
    int id = blockIdx.x * blockDim.x + threadIdx.x;
    int total = OT_TOTAL * C * 32 * 16;
    if (id >= total) return;
    int h    = id & 15;
    int lane = (id >> 4) & 31;
    int c    = (id >> 9) % C;
    int ot   = id / (C << 9);
    int t = ((lane >> 4) << 4) + h;           // logical K within padded channel
    int o = ot * 16 + (lane & 15);            // N = lane & 15
    float v = (t < 25) ? src[(o * C + c) * 25 + t] : 0.0f;
    dst[id] = (_Float16)v;
}

// ---------------------------------------------------------------------------
// Fused INRF layer: out = conv5(x, m) + b - 2 * sum_k w(k) tanh((x_k - x_c)/s)
// One block = 6 waves = one 16-pixel M tile, covering all 12 N tiles.
// A-fragments (raw patch + tanh patch) are built cooperatively in LDS in WMMA
// A-layout (branch-free, b128 stores), then consumed by all waves via
// ds_load_b128 + v_wmma_f32_16x16x32_f16 chains (f32 accumulation).
// ---------------------------------------------------------------------------
template<int H, int W, int CIN>
__global__ __launch_bounds__(192) void inrf_wmma(
    const float* __restrict__ x,
    const _Float16* __restrict__ pm,   // packed m weights (B-fragments)
    const _Float16* __restrict__ pw,   // packed w weights (B-fragments)
    const float* __restrict__ mb,      // bias
    float* __restrict__ out)
{
    constexpr int TILES  = (H * W) / 16;        // 16-pixel M tiles per image
    constexpr int CH     = (CIN < 16) ? CIN : 16;   // channels per LDS chunk
    constexpr int NCHUNK = CIN / CH;
    __shared__ __align__(32) _Float16 amf[CH * 512];  // [c][lane][16h] raw patch
    __shared__ __align__(32) _Float16 asf[CH * 512];  // [c][lane][16h] tanh patch

    const int b        = blockIdx.x / TILES;
    const int t16      = blockIdx.x % TILES;
    const int pix_base = t16 * 16;

    const int tid    = threadIdx.x;
    const int lane   = tid & 31;
    const int laneLo = lane & 15;
    const int laneHi = lane >> 4;
    const int ot0    = (tid >> 5) * 2;          // 2 N-tiles per wave, 6 waves

    v8f accm0 = {}, accs0 = {}, accm1 = {}, accs1 = {};

    for (int ch = 0; ch < NCHUNK; ++ch) {
        const int cbase = ch * CH;

        // ---- cooperative A-fragment build: 8 halfs (one b128) per item ----
        for (int i = tid; i < CH * 64; i += 192) {
            int g  = i & 1;                    // half-group within lane frag
            int ln = (i >> 1) & 31;
            int cl = i >> 6;
            int px = ln & 15;                  // A-row pixel
            int pr = (pix_base + px) / W;
            int pc = (pix_base + px) % W;
            const float* xc = x + (b * CIN + cbase + cl) * (H * W);
            float center = xc[pr * W + pc];
            int tb = (ln >> 4) * 16 + g * 8;   // A-layout K base for this group (contiguous k = 16*half + s)
            v8h vm, vs;
#pragma unroll
            for (int j = 0; j < 8; ++j) {
                int t  = tb + j;
                int gh = pr + t / 5 - 2;
                int gw = pc + t % 5 - 2;
                bool inb = (t < 25) & (gh >= 0) & (gh < H) & (gw >= 0) & (gw < W);
                int idx = inb ? (gh * W + gw) : 0;   // clamp, no branch
                float pv = xc[idx];
                pv = inb ? pv : 0.0f;                // select, no branch
                vm[j] = (_Float16)pv;
                vs[j] = (_Float16)fast_tanh((pv - center) * 12.5f);
            }
            int off = cl * 512 + ln * 16 + g * 8;
            *(v8h*)(amf + off) = vm;
            *(v8h*)(asf + off) = vs;
        }
        __syncthreads();

        // ---- consume: 4 WMMAs per channel (2 o-tiles x {m, tanh}) ----
        for (int cl = 0; cl < CH; ++cl) {
            int c = cbase + cl;
            v16h am = *(const v16h*)(amf + cl * 512 + lane * 16);
            v16h as = *(const v16h*)(asf + cl * 512 + lane * 16);
            const v16h* bmp = (const v16h*)pm + ((ot0 * CIN + c) * 32 + lane);
            const v16h* bwp = (const v16h*)pw + ((ot0 * CIN + c) * 32 + lane);
            v16h bm0 = bmp[0];
            v16h bw0 = bwp[0];
            v16h bm1 = bmp[CIN * 32];
            v16h bw1 = bwp[CIN * 32];
            accm0 = __builtin_amdgcn_wmma_f32_16x16x32_f16(false, am, false, bm0,
                                                           (short)0, accm0, false, false);
            accs0 = __builtin_amdgcn_wmma_f32_16x16x32_f16(false, as, false, bw0,
                                                           (short)0, accs0, false, false);
            accm1 = __builtin_amdgcn_wmma_f32_16x16x32_f16(false, am, false, bm1,
                                                           (short)0, accm1, false, false);
            accs1 = __builtin_amdgcn_wmma_f32_16x16x32_f16(false, as, false, bw1,
                                                           (short)0, accs1, false, false);
        }
        __syncthreads();
    }

    // ---- epilogue: D layout -> M = v + 8*laneHi (pixel), N = laneLo ----
#pragma unroll
    for (int which = 0; which < 2; ++which) {
        int o = (ot0 + which) * 16 + laneLo;
        float bias = mb[o];
        v8f am_ = which ? accm1 : accm0;
        v8f as_ = which ? accs1 : accs0;
#pragma unroll
        for (int v = 0; v < 8; ++v) {
            int pix = pix_base + v + 8 * laneHi;
            int gh = pix / W;
            int gw = pix % W;
            out[((b * NOUT + o) * H + gh) * W + gw] = am_[v] + bias - 2.0f * as_[v];
        }
    }
}

// ---------------------------------------------------------------------------
__global__ void maxpool2(const float* __restrict__ in, float* __restrict__ out,
                         int total, int OH, int OW) {
    int i = blockIdx.x * blockDim.x + threadIdx.x;
    if (i >= total) return;
    int ow = i % OW;
    int oh = (i / OW) % OH;
    int bc = i / (OW * OH);
    const float* p = in + ((bc * (OH * 2)) + oh * 2) * (OW * 2) + ow * 2;
    float a = fmaxf(p[0], p[1]);
    float q = fmaxf(p[OW * 2], p[OW * 2 + 1]);
    out[i] = fmaxf(a, q);
}

// avgpool8 + FC head: one block per batch image
__global__ __launch_bounds__(192) void head_kernel(
    const float* __restrict__ x4, const float* __restrict__ fcw,
    const float* __restrict__ fcb, float* __restrict__ logits) {
    __shared__ float feat[NOUT];
    int b = blockIdx.x;
    int c = threadIdx.x;
    const float* p = x4 + (b * NOUT + c) * 64;
    float s = 0.0f;
    for (int i = 0; i < 64; ++i) s += p[i];
    feat[c] = s * (1.0f / 64.0f);
    __syncthreads();
    if (c < 10) {
        float acc = fcb[c];
        for (int k = 0; k < NOUT; ++k) acc += fcw[c * NOUT + k] * feat[k];
        logits[b * 10 + c] = acc;
    }
}

// ---------------------------------------------------------------------------
extern "C" void kernel_launch(void* const* d_in, const int* in_sizes, int n_in,
                              void* d_out, int out_size, void* d_ws, size_t ws_size,
                              hipStream_t stream) {
    const float* x    = (const float*)d_in[0];
    const float* m1_w = (const float*)d_in[1];
    const float* m1_b = (const float*)d_in[2];
    const float* w1_w = (const float*)d_in[3];
    const float* m2_w = (const float*)d_in[4];
    const float* m2_b = (const float*)d_in[5];
    const float* w2_w = (const float*)d_in[6];
    const float* m3_w = (const float*)d_in[7];
    const float* m3_b = (const float*)d_in[8];
    const float* w3_w = (const float*)d_in[9];
    const float* fc_w = (const float*)d_in[10];
    const float* fc_b = (const float*)d_in[11];

    char* ws = (char*)d_ws;
    float*    y1  = (float*)(ws);                   // [64,192,32,32] 50.33MB
    float*    x03 = (float*)(ws + 50331648);        // [64,192,8,8]
    float*    x04 = (float*)(ws + 53477376);        // [64,192,8,8]
    _Float16* pm1 = (_Float16*)(ws + 56623104);     // 36KB
    _Float16* pw1 = (_Float16*)(ws + 56659968);
    _Float16* pm2 = (_Float16*)(ws + 56696832);     // 2.25MB each
    _Float16* pw2 = (_Float16*)(ws + 59056128);
    _Float16* pm3 = (_Float16*)(ws + 61415424);
    _Float16* pw3 = (_Float16*)(ws + 63774720);

    float* logits = (float*)d_out;
    float* x00 = logits + 640;                      // [64,192,16,16]
    float* x01 = x00 + 64 * 192 * 16 * 16;          // [64,192,16,16]

    // Pack all conv/w weights into f16 WMMA B-fragment layout (K padded to 32)
    int tot1 = OT_TOTAL * 3 * 512;
    pack_wmma_b<<<(tot1 + 255) / 256, 256, 0, stream>>>(m1_w, pm1, 3);
    pack_wmma_b<<<(tot1 + 255) / 256, 256, 0, stream>>>(w1_w, pw1, 3);
    int tot2 = OT_TOTAL * 192 * 512;
    pack_wmma_b<<<(tot2 + 255) / 256, 256, 0, stream>>>(m2_w, pm2, 192);
    pack_wmma_b<<<(tot2 + 255) / 256, 256, 0, stream>>>(w2_w, pw2, 192);
    pack_wmma_b<<<(tot2 + 255) / 256, 256, 0, stream>>>(m3_w, pm3, 192);
    pack_wmma_b<<<(tot2 + 255) / 256, 256, 0, stream>>>(w3_w, pw3, 192);

    // Layer 1: INRF(3->192) on 32x32, then pool -> x00 (net output #2)
    inrf_wmma<32, 32, 3><<<64 * 64, 192, 0, stream>>>(x, pm1, pw1, m1_b, y1);
    maxpool2<<<(64 * 192 * 16 * 16 + 255) / 256, 256, 0, stream>>>(
        y1, x00, 64 * 192 * 16 * 16, 16, 16);

    // Layer 2: INRF(192->192) on 16x16 -> x01 (net output #3), then pool
    inrf_wmma<16, 16, 192><<<64 * 16, 192, 0, stream>>>(x00, pm2, pw2, m2_b, x01);
    maxpool2<<<(64 * 192 * 8 * 8 + 255) / 256, 256, 0, stream>>>(
        x01, x03, 64 * 192 * 8 * 8, 8, 8);

    // Layer 3: INRF(192->192) on 8x8
    inrf_wmma<8, 8, 192><<<64 * 4, 192, 0, stream>>>(x03, pm3, pw3, m3_b, x04);

    // avgpool8 + FC
    head_kernel<<<64, 192, 0, stream>>>(x04, fc_w, fc_b, logits);
}
